// CoordinateDecoder_18021682774655
// MI455X (gfx1250) — matrix operand order repaired and checked by reading, not compiled, not run on hardware
//
#include <hip/hip_runtime.h>
#include <hip/hip_bf16.h>
#include <stdint.h>

typedef __attribute__((ext_vector_type(16))) _Float16 v16h;
typedef __attribute__((ext_vector_type(8)))  float    v8f;

#define NFREQ   10
#define IMGB    8
#define IMGH    64
#define IMGW    64
#define IMGC    256
#define NPTS    65536
#define WIDTH   128
#define ROWS    128            // rows per block
#define TILES   (NPTS / ROWS)  // 512
#define XK      320            // padded layer-0 K (enc 0..41, zero 42..47, feat 48..303, zero 304..319)
#define XSTRIDE 328            // LDS stride (halves), conflict-avoiding
#define HSTRIDE 136

// Packed weights: each 32x16 f16 B-fragment block = 256 u32 words.
// Word layout: blk*256 + lane*8 + v  -> per-lane contiguous = 2x b128 loads.
#define BLK_W0     0           // 10 ktiles x 8 ntiles
#define BLK_W1     80          // 4 x 8
#define BLK_W2     112         // 4 x 8
#define BLK_WO     144         // 4 x 1
#define TOTAL_BLKS 148
#define PACK_WORDS (TOTAL_BLKS * 256)          // 37888 u32
#define GRID_BYTE_OFF ((size_t)PACK_WORDS * 4) // 151552 B (16B aligned)
#define GRID_ELEMS (IMGB * IMGH * IMGW * IMGC) // 8388608

// ---------------------------------------------------------------------------
// Prep 1: pack weights into WMMA B-fragment order, f32 -> f16.
// B layout used here (contiguous k, not ISA table order): lane n<16 holds column N=n, VGPR v holds K=2v,2v+1;
// lanes 16..31 hold same columns at K+16.
// ---------------------------------------------------------------------------
__global__ void pack_weights_kernel(const float* __restrict__ w0,
                                    const float* __restrict__ w1,
                                    const float* __restrict__ w2,
                                    const float* __restrict__ wo,
                                    uint32_t* __restrict__ wp) {
  int tid = blockIdx.x * blockDim.x + threadIdx.x;
  if (tid >= PACK_WORDS) return;
  const int v    = tid & 7;
  const int lane = (tid >> 3) & 31;
  const int blk  = tid >> 8;
  const int rsub = lane & 15;
  const int hi   = lane >> 4;
  const int kin  = v * 2 + hi * 16;   // K within 32-wide tile (even)

  float lo = 0.f, hif = 0.f;
  if (blk < BLK_W1) {                                  // W0: padded K remap
    int kt = blk >> 3, nt = blk & 7;
    int n = nt * 16 + rsub;
    int k = kt * 32 + kin;
    #pragma unroll
    for (int s = 0; s < 2; ++s) {
      int kk = k + s;
      int row = (kk < 42) ? kk : ((kk >= 48 && kk < 304) ? kk - 6 : -1);
      float val = (row >= 0) ? w0[row * WIDTH + n] : 0.f;
      if (s == 0) lo = val; else hif = val;
    }
  } else if (blk < BLK_WO) {                           // W1 / W2
    int b2 = blk - BLK_W1;
    const float* w = (b2 < 32) ? w1 : w2;
    if (b2 >= 32) b2 -= 32;
    int kt = b2 >> 3, nt = b2 & 7;
    int n = nt * 16 + rsub;
    int k = kt * 32 + kin;
    lo  = w[k * WIDTH + n];
    hif = w[(k + 1) * WIDTH + n];
  } else {                                             // W_out: [128,3], pad N to 16
    int kt = blk - BLK_WO;
    int n = rsub;
    int k = kt * 32 + kin;
    lo  = (n < 3) ? wo[k * 3 + n] : 0.f;
    hif = (n < 3) ? wo[(k + 1) * 3 + n] : 0.f;
  }
  union { _Float16 h[2]; uint32_t u; } p;
  p.h[0] = (_Float16)lo; p.h[1] = (_Float16)hif;
  wp[tid] = p.u;
}

// ---------------------------------------------------------------------------
// Prep 2: feature grid f32 -> f16 (halves L2 gather bytes; grid fits in L2)
// ---------------------------------------------------------------------------
__global__ void convert_grid_kernel(const float* __restrict__ g,
                                    _Float16* __restrict__ gh) {
  int i = blockIdx.x * blockDim.x + threadIdx.x;
  if (i < GRID_ELEMS) gh[i] = (_Float16)g[i];
}

// ---------------------------------------------------------------------------
// Fragment loaders
// ---------------------------------------------------------------------------
// A 16x32 f16 fragment from LDS (contiguous k order, same as packed B):
// lane m<16 = row m; VGPR v holds K = v*2 + hi*16 (and +1).
// (dword groups are contiguous -> compiler emits 2x ds_load_b128)
__device__ __forceinline__ v16h load_a_frag(const _Float16* base, int row0,
                                            int stride, int kb, int rsub, int hi) {
  union { uint32_t u[8]; v16h v; } f;
  const char* p = (const char*)(base + (row0 + rsub) * stride + kb);
  #pragma unroll
  for (int vv = 0; vv < 8; ++vv) {
    int k = vv * 2 + hi * 16;
    f.u[vv] = *(const uint32_t*)(p + k * 2);
  }
  return f.v;
}

// B fragment: per-lane contiguous packed words -> 2x global_load_b128.
__device__ __forceinline__ v16h load_b_frag(const uint32_t* __restrict__ wp,
                                            int blk, int lane) {
  union { uint4 q[2]; v16h v; } f;
  const uint4* p = (const uint4*)(wp + blk * 256 + lane * 8);
  f.q[0] = p[0];
  f.q[1] = p[1];
  return f.v;
}

// tanh via hardware exp (v_exp_f32): tanh(z) = (e-1)/(e+1), e = exp(2z).
// Clamp keeps exp finite; tanh is fully saturated at |z|=10 anyway.
__device__ __forceinline__ float fast_tanh(float z) {
  z = fminf(fmaxf(z, -10.f), 10.f);
  float e = __expf(2.f * z);
  return (e - 1.f) * __frcp_rn(e + 1.f);
}

__device__ __forceinline__ float gelu_f(float x) {
  return 0.5f * x * (1.f + fast_tanh(0.7978845608f * (x + 0.044715f * x * x * x)));
}

// ---------------------------------------------------------------------------
// Main fused kernel: encode + bilinear gather + 4-layer WMMA MLP
// ---------------------------------------------------------------------------
__global__ __launch_bounds__(256) void decoder_kernel(
    const float* __restrict__ coords, const _Float16* __restrict__ gridh,
    const uint32_t* __restrict__ wp,
    const float* __restrict__ b0, const float* __restrict__ b1,
    const float* __restrict__ b2, const float* __restrict__ bo,
    float* __restrict__ out) {
  extern __shared__ __align__(16) char smem[];
  _Float16* Xin = (_Float16*)smem;                                  // ROWS x XSTRIDE
  _Float16* Hb  = (_Float16*)(smem + ROWS * XSTRIDE * 2);           // ROWS x HSTRIDE
  float4*   samp = (float4*)(smem + ROWS * XSTRIDE * 2 + ROWS * HSTRIDE * 2);

  const int t    = threadIdx.x;
  const int lane = t & 31;
  const int wv   = t >> 5;          // wave id 0..7, owns rows [wv*16, wv*16+16)
  const int rsub = lane & 15;
  const int hi   = lane >> 4;
  const int img  = blockIdx.x >> 9;
  const int tile = blockIdx.x & (TILES - 1);
  const int row0 = tile * ROWS;

  // ---- Phase 1: positional encoding + sampling params (one thread per row)
  if (t < ROWS) {
    const int n = row0 + t;
    const float cx = coords[2 * n + 0];
    const float cy = coords[2 * n + 1];
    _Float16* xr = Xin + t * XSTRIDE;
    xr[0] = (_Float16)cx;
    xr[1] = (_Float16)cy;
    float fr = 3.14159265358979f;
    #pragma unroll
    for (int f = 0; f < NFREQ; ++f) {
      float ax = cx * fr, ay = cy * fr;
      xr[2 + 4 * f + 0] = (_Float16)__sinf(ax);
      xr[2 + 4 * f + 1] = (_Float16)__sinf(ay);
      xr[2 + 4 * f + 2] = (_Float16)__cosf(ax);
      xr[2 + 4 * f + 3] = (_Float16)__cosf(ay);
      fr *= 2.f;
    }
    #pragma unroll
    for (int k = 42; k < 48; ++k) xr[k] = (_Float16)0.f;
    #pragma unroll
    for (int k = 304; k < XK; ++k) xr[k] = (_Float16)0.f;
    // pixel coords: row from coords[:,0], col from coords[:,1]
    float py = (cx + 1.f) * 0.5f * (float)(IMGH - 1);
    float px = (cy + 1.f) * 0.5f * (float)(IMGW - 1);
    py = fminf(fmaxf(py, 0.f), (float)(IMGH - 1));
    px = fminf(fmaxf(px, 0.f), (float)(IMGW - 1));
    float y0 = floorf(py), x0 = floorf(px);
    samp[t] = make_float4(y0, x0, py - y0, px - x0);
  }
  __syncthreads();

  // ---- Phase 2: bilinear gather, 256 ch/row; each wave does one row/iter,
  // each lane 8 contiguous channels -> fully coalesced 512B per corner.
  {
    const _Float16* gb = gridh + (size_t)img * IMGH * IMGW * IMGC;
    const int c = lane * 8;
    #pragma unroll 2
    for (int it = 0; it < ROWS / 8; ++it) {
      const int r = wv + it * 8;
      float4 s = samp[r];
      int y0 = (int)s.x, x0 = (int)s.y;
      float fy = s.z, fx = s.w;
      int y1 = min(y0 + 1, IMGH - 1), x1 = min(x0 + 1, IMGW - 1);
      union U4 { uint4 q; _Float16 h[8]; } a00, a01, a10, a11, o;
      a00.q = *(const uint4*)(gb + ((y0 * IMGW + x0) * IMGC + c));
      a01.q = *(const uint4*)(gb + ((y0 * IMGW + x1) * IMGC + c));
      a10.q = *(const uint4*)(gb + ((y1 * IMGW + x0) * IMGC + c));
      a11.q = *(const uint4*)(gb + ((y1 * IMGW + x1) * IMGC + c));
      float w00 = (1.f - fy) * (1.f - fx), w01 = (1.f - fy) * fx;
      float w10 = fy * (1.f - fx),         w11 = fy * fx;
      #pragma unroll
      for (int j = 0; j < 8; ++j)
        o.h[j] = (_Float16)(w00 * (float)a00.h[j] + w01 * (float)a01.h[j] +
                            w10 * (float)a10.h[j] + w11 * (float)a11.h[j]);
      *(uint4*)&Xin[r * XSTRIDE + 48 + c] = o.q;
    }
  }
  __syncthreads();

  // ---- Layer 0: [16 x 320] @ [320 x 128], K tiles = 10 (fully unrolled;
  // all 8 B fragments preloaded per ktile so waits amortize 8 deep)
  v8f acc[8] = {};
  #pragma unroll
  for (int kt = 0; kt < 10; ++kt) {
    v16h a = load_a_frag(Xin, wv * 16, XSTRIDE, kt * 32, rsub, hi);
    v16h bf[8];
    #pragma unroll
    for (int nt = 0; nt < 8; ++nt)
      bf[nt] = load_b_frag(wp, BLK_W0 + kt * 8 + nt, lane);
    #pragma unroll
    for (int nt = 0; nt < 8; ++nt)
      acc[nt] = __builtin_amdgcn_wmma_f32_16x16x32_f16(
          false, a, false, bf[nt], (short)0, acc[nt], false, false);
  }
  #pragma unroll
  for (int nt = 0; nt < 8; ++nt) {
    float bias = b0[nt * 16 + rsub];
    #pragma unroll
    for (int j = 0; j < 8; ++j) {
      float g = gelu_f(acc[nt][j] + bias);
      Hb[(wv * 16 + j + 8 * hi) * HSTRIDE + nt * 16 + rsub] = (_Float16)g;
    }
  }
  // No barrier: each wave reads back only the strip it wrote (in-order DS).

  // ---- Layers 1 and 2: [16 x 128] @ [128 x 128], in-place per-wave strip
  #pragma unroll 1
  for (int layer = 0; layer < 2; ++layer) {
    const int blkbase = (layer == 0) ? BLK_W1 : BLK_W2;
    const float* bias_p = (layer == 0) ? b1 : b2;
    #pragma unroll
    for (int nt = 0; nt < 8; ++nt)
      acc[nt] = (v8f){0.f, 0.f, 0.f, 0.f, 0.f, 0.f, 0.f, 0.f};
    #pragma unroll
    for (int kt = 0; kt < 4; ++kt) {
      v16h a = load_a_frag(Hb, wv * 16, HSTRIDE, kt * 32, rsub, hi);
      v16h bf[8];
      #pragma unroll
      for (int nt = 0; nt < 8; ++nt)
        bf[nt] = load_b_frag(wp, blkbase + kt * 8 + nt, lane);
      #pragma unroll
      for (int nt = 0; nt < 8; ++nt)
        acc[nt] = __builtin_amdgcn_wmma_f32_16x16x32_f16(
            false, a, false, bf[nt], (short)0, acc[nt], false, false);
    }
    #pragma unroll
    for (int nt = 0; nt < 8; ++nt) {
      float bias = bias_p[nt * 16 + rsub];
      #pragma unroll
      for (int j = 0; j < 8; ++j) {
        float g = gelu_f(acc[nt][j] + bias);
        Hb[(wv * 16 + j + 8 * hi) * HSTRIDE + nt * 16 + rsub] = (_Float16)g;
      }
    }
  }

  // ---- Output layer: [16 x 128] @ [128 x 16(pad of 3)], tanh
  v8f acco = {};
  #pragma unroll
  for (int kt = 0; kt < 4; ++kt) {
    v16h a = load_a_frag(Hb, wv * 16, HSTRIDE, kt * 32, rsub, hi);
    v16h bb = load_b_frag(wp, BLK_WO + kt, lane);
    acco = __builtin_amdgcn_wmma_f32_16x16x32_f16(
        false, a, false, bb, (short)0, acco, false, false);
  }
  if (rsub < 3) {
    float bias = bo[rsub];
    #pragma unroll
    for (int j = 0; j < 8; ++j) {
      int m = wv * 16 + j + 8 * hi;
      out[((size_t)img * NPTS + row0 + m) * 3 + rsub] = fast_tanh(acco[j] + bias);
    }
  }
}

// ---------------------------------------------------------------------------
extern "C" void kernel_launch(void* const* d_in, const int* in_sizes, int n_in,
                              void* d_out, int out_size, void* d_ws, size_t ws_size,
                              hipStream_t stream) {
  const float* feature_grid = (const float*)d_in[0];
  const float* coords       = (const float*)d_in[1];
  const float* w0   = (const float*)d_in[2];
  const float* b0   = (const float*)d_in[3];
  const float* w1   = (const float*)d_in[4];
  const float* b1   = (const float*)d_in[5];
  const float* w2   = (const float*)d_in[6];
  const float* b2   = (const float*)d_in[7];
  const float* wo   = (const float*)d_in[8];
  const float* bo   = (const float*)d_in[9];

  uint32_t* wp      = (uint32_t*)d_ws;                          // 151,552 B
  _Float16* gridh   = (_Float16*)((char*)d_ws + GRID_BYTE_OFF); // 16,777,216 B

  pack_weights_kernel<<<(PACK_WORDS + 255) / 256, 256, 0, stream>>>(w0, w1, w2, wo, wp);
  convert_grid_kernel<<<(GRID_ELEMS + 255) / 256, 256, 0, stream>>>(feature_grid, gridh);

  size_t smem = (size_t)ROWS * XSTRIDE * 2 + (size_t)ROWS * HSTRIDE * 2 + (size_t)ROWS * 16;
  decoder_kernel<<<dim3(IMGB * TILES), dim3(256), smem, stream>>>(
      coords, gridh, wp, b0, b1, b2, bo, (float*)d_out);
}
